// AnchorStore_13426067767648
// MI455X (gfx1250) — matrix operand order repaired and checked by reading, not compiled, not run on hardware
//
#include <hip/hip_runtime.h>

// ---------------------------------------------------------------------------
// MI455X (gfx1250) KNN-calibrate:
//   scaled[b,k] = (sum_d a[k,d]*log x[b,d] - sum_d a[k,d]*log a[k,d]) / (D*T)
//   out[b,c]    = softmax over top-knn scaled, scattered by queue_label
//
// Roofline: anchors (4096x50257 fp32 = 823 MB) stream ONCE from HBM
// (~35 us @ 23.3 TB/s) with non-temporal hints so the bf16 log(logits)
// (12.9 MB) stays L2-resident for all 256 k-tile workgroups. The 52.7 GFLOP
// GEMM runs on v_wmma_f32_16x16x32_bf16 (fp32 accumulate), so compute sits
// well under the memory roofline. Anchor chunks are staged into LDS once per
// workgroup via global_load_async_to_lds_b32 (ASYNCcnt, double-buffered) to
// kill the 8x per-wave load redundancy.
// ---------------------------------------------------------------------------

typedef __attribute__((ext_vector_type(16))) __bf16 v16bf;
typedef __attribute__((ext_vector_type(8)))  __bf16 v8bf;
typedef __attribute__((ext_vector_type(8)))  float  v8f;

#define KNN_T 0.05f

// ---------------------------------------------------------------------------
// Kernel 1: xlog[b, 0..Dp) = bf16(log(logits[b, d])), zero in pad region.
// ---------------------------------------------------------------------------
__global__ __launch_bounds__(256) void log_convert_kernel(
    const float* __restrict__ logits, __bf16* __restrict__ xlog,
    int B, int D, int Dp)
{
    long long t = (long long)blockIdx.x * blockDim.x + threadIdx.x;
    long long total = (long long)B * Dp;
    if (t >= total) return;
    int d = (int)(t % Dp);
    long long b = t / Dp;
    float v = 0.0f;
    if (d < D) v = __logf(logits[b * D + d]);
    xlog[t] = (__bf16)v;
}

// ---------------------------------------------------------------------------
// Kernel 2: fused WMMA GEMM + self-term.
//   grid.x = K/16 k-tiles, 256 threads = 8 waves; wave w owns the 16x16
//   output tile rows [bt+16w, bt+16w+16) x cols [k0, k0+16).
//   Anchor chunks (16 rows x 32 d, fp32) are async-staged into LDS once per
//   block (double-buffered, 34-dword row stride for conflict-free reads).
// ---------------------------------------------------------------------------
#define BROW_STRIDE 34            // LDS dwords per staged anchor row
#define CHUNK_DW    (16 * BROW_STRIDE)

__global__ __launch_bounds__(256) void knn_gemm_kernel(
    const __bf16* __restrict__ xlog,   // [B][Dp] bf16 (padded, pad = 0)
    const float*  __restrict__ anchor, // [K][D]  fp32
    float*        __restrict__ scaled, // [B][K]  fp32 output
    int B, int D, int Dp, int K)
{
    __shared__ float bstage[2][CHUNK_DW];   // double-buffered anchor chunk
    __shared__ float lds_self[16];

    const int tid  = threadIdx.x;
    const int wave = tid >> 5;
    const int lane = tid & 31;
    const int half = lane >> 4;   // 0: lanes 0-15, 1: lanes 16-31
    const int n    = lane & 15;   // A-row (M) and B-column (N) for this lane
    const int k0   = blockIdx.x * 16;
    const int bt   = blockIdx.y * 128;

    if (tid < 16) lds_self[tid] = 0.0f;

    // ---- async staging assignment: thread stages chunk dwords tid, tid+256
    //      chunk dword q -> row q>>5 (0..15), d-offset q&31
    const int r0  = tid >> 5;          // rows 0..7
    const int r1  = r0 + 8;            // rows 8..15
    const int dd  = tid & 31;
    int kr0 = k0 + r0; kr0 = kr0 < K ? kr0 : (K - 1);
    int kr1 = k0 + r1; kr1 = kr1 < K ? kr1 : (K - 1);
    const float* g0 = anchor + (size_t)kr0 * D + dd;
    const float* g1 = anchor + (size_t)kr1 * D + dd;
    const unsigned l0 = (unsigned)(size_t)&bstage[0][r0 * BROW_STRIDE + dd];
    const unsigned l1 = (unsigned)(size_t)&bstage[0][r1 * BROW_STRIDE + dd];
    const unsigned bufbytes = (unsigned)(CHUNK_DW * sizeof(float));

#define STAGE_CHUNK(buf, d0)                                                   \
    do {                                                                       \
        unsigned _dst0 = l0 + (unsigned)(buf) * bufbytes;                      \
        unsigned _dst1 = l1 + (unsigned)(buf) * bufbytes;                      \
        const float* _s0 = g0 + (d0);                                          \
        const float* _s1 = g1 + (d0);                                          \
        asm volatile("global_load_async_to_lds_b32 %0, %1, off th:TH_LOAD_NT"  \
                     :: "v"(_dst0), "v"(_s0) : "memory");                      \
        asm volatile("global_load_async_to_lds_b32 %0, %1, off th:TH_LOAD_NT"  \
                     :: "v"(_dst1), "v"(_s1) : "memory");                      \
    } while (0)

    // ---- A operand pointer (unique per wave): row bt + wave*16 + n
    int arow = bt + wave * 16 + n;
    int arow_c = arow < B ? arow : (B - 1);
    // elems 0..7 -> K = 8*half.., elems 8..15 -> K = 16 + 8*half..
    const __bf16* ap = xlog + (size_t)arow_c * Dp + 8 * half;

    const int krow = k0 + n;
    const int M = D / 32;              // full 32-wide chunks (1570)

    v8f acc = {};
    float st = 0.0f;                   // partial sum of a*log(a)

    v8bf a0c = {}, a1c = {};
    if (M > 0) {
        STAGE_CHUNK(0, 0);
        a0c = *(const v8bf*)(ap);
        a1c = *(const v8bf*)(ap + 16);
    }

    for (int it = 0; it < M; ++it) {
        // software pipeline: issue next chunk (LDS async + A regs), then wait
        v8bf a0n = {}, a1n = {};
        if (it + 1 < M) {
            const int dn = (it + 1) * 32;
            a0n = *(const v8bf*)(ap + dn);
            a1n = *(const v8bf*)(ap + dn + 16);
            STAGE_CHUNK((it + 1) & 1, dn);
            asm volatile("s_wait_asynccnt 0x2" ::: "memory");
        } else {
            asm volatile("s_wait_asynccnt 0x0" ::: "memory");
        }
        __syncthreads();               // chunk `it` fully staged

        // ---- consume chunk it: lane reads 16 fp32 (row n, d 8*half..+8, 16+8*half..+8)
        const float* src = &bstage[it & 1][n * BROW_STRIDE + 8 * half];
        float bv[16];
        #pragma unroll
        for (int e = 0; e < 8; ++e) {
            float2 t = *(const float2*)(src + 2 * e + (e >= 4 ? 8 : 0));   // 8B aligned, no conflicts
            bv[2 * e] = t.x; bv[2 * e + 1] = t.y;
        }

        // fused self-term: each wave owns chunks it%8==wave (exactly-once)
        if ((it & 7) == wave) {
            #pragma unroll
            for (int e = 0; e < 16; ++e)
                st += (bv[e] > 0.0f) ? bv[e] * __logf(bv[e]) : 0.0f;
        }

        v16bf a, b;
        #pragma unroll
        for (int e = 0; e < 8; ++e) { a[e] = a0c[e]; a[8 + e] = a1c[e]; }
        #pragma unroll
        for (int e = 0; e < 16; ++e) b[e] = (__bf16)bv[e];

        acc = __builtin_amdgcn_wmma_f32_16x16x32_bf16(
            false, a, false, b, (short)0, acc, false, false);

        __syncthreads();               // all waves done reading before restage
        a0c = a0n; a1c = a1n;
    }

    // ---- epilogue: remaining D - M*32 elements (17), guarded loads
    if (D > M * 32) {
        const int d0 = M * 32;
        v16bf a, b;
        v8bf t0 = *(const v8bf*)(ap + d0);          // xlog pad region is 0
        v8bf t1 = *(const v8bf*)(ap + d0 + 16);
        #pragma unroll
        for (int e = 0; e < 8; ++e) { a[e] = t0[e]; a[8 + e] = t1[e]; }

        const int krow_c = krow < K ? krow : (K - 1);
        const float* bptr = anchor + (size_t)krow_c * D;
        float bv[16];
        #pragma unroll
        for (int e = 0; e < 16; ++e) {
            int d  = d0 + 8 * half + e + (e >= 8 ? 8 : 0);
            int dc = d < D ? d : (D - 1);
            float v = __builtin_nontemporal_load(bptr + dc);
            bv[e] = (d < D) ? v : 0.0f;
        }
        if (wave == 0) {
            #pragma unroll
            for (int e = 0; e < 16; ++e)
                st += (bv[e] > 0.0f) ? bv[e] * __logf(bv[e]) : 0.0f;
        }
        #pragma unroll
        for (int e = 0; e < 16; ++e) b[e] = (__bf16)bv[e];
        acc = __builtin_amdgcn_wmma_f32_16x16x32_bf16(
            false, a, false, b, (short)0, acc, false, false);
    }

    // ---- reduce self-term: lane j / j+16 of all waves own row k0+j
    atomicAdd(&lds_self[n], st);       // ds_add_f32
    __syncthreads();

    const float self = lds_self[n];
    const float inv  = 1.0f / ((float)D * KNN_T);

    // C layout: VGPR r -> (m = r + 8*half, n = lane&15)
    #pragma unroll
    for (int r = 0; r < 8; ++r) {
        int gb = bt + wave * 16 + r + 8 * half;
        if (gb < B && krow < K)
            scaled[(size_t)gb * K + krow] = (acc[r] - self) * inv;
    }
#undef STAGE_CHUNK
}

// ---------------------------------------------------------------------------
// Kernel 3: per-row top-knn + softmax + label scatter. One block per b-row.
// ---------------------------------------------------------------------------
__global__ __launch_bounds__(256) void topk_kernel(
    const float* __restrict__ scaled, const int* __restrict__ labels,
    const int* __restrict__ knn_p, const int* __restrict__ nclass_p,
    float* __restrict__ out, int K)
{
    __shared__ float vals[4096];
    __shared__ float rv[256];
    __shared__ int   ri[256];
    __shared__ float topd[32];
    __shared__ int   topi[32];

    const int b   = blockIdx.x;
    const int tid = threadIdx.x;
    int knn = knn_p[0];     knn = knn < 1 ? 1 : (knn > 32 ? 32 : knn);
    int nc  = nclass_p[0];  nc  = nc  < 1 ? 1 : (nc  > 32 ? 32 : nc);
    int Kc  = K < 4096 ? K : 4096;

    const float* row = scaled + (size_t)b * K;
    for (int i = tid; i < Kc; i += 256) vals[i] = row[i];
    __syncthreads();

    for (int s = 0; s < knn; ++s) {
        float mv = -__builtin_inff();
        int   mi = 0x7fffffff;
        for (int i = tid; i < Kc; i += 256) {
            float v = vals[i];
            if (v > mv || (v == mv && i < mi)) { mv = v; mi = i; }
        }
        rv[tid] = mv; ri[tid] = mi;
        __syncthreads();
        for (int off = 128; off > 0; off >>= 1) {
            if (tid < off) {
                float ov = rv[tid + off]; int oi = ri[tid + off];
                if (ov > rv[tid] || (ov == rv[tid] && oi < ri[tid])) {
                    rv[tid] = ov; ri[tid] = oi;
                }
            }
            __syncthreads();
        }
        if (tid == 0) {
            topd[s] = rv[0]; topi[s] = ri[0];
            vals[ri[0]] = -__builtin_inff();
        }
        __syncthreads();
    }

    if (tid == 0) {
        float m = topd[0];
        for (int j = 1; j < knn; ++j) m = fmaxf(m, topd[j]);
        float w[32], sum = 0.0f;
        for (int j = 0; j < knn; ++j) { w[j] = __expf(topd[j] - m); sum += w[j]; }
        float cls[32];
        for (int c = 0; c < nc; ++c) cls[c] = 0.0f;
        for (int j = 0; j < knn; ++j) {
            int l = labels[topi[j]];
            if (l >= 0 && l < nc) cls[l] += w[j];
        }
        float invs = 1.0f / sum;
        for (int c = 0; c < nc; ++c) out[(size_t)b * nc + c] = cls[c] * invs;
    }
}

// ---------------------------------------------------------------------------
// Launch: inputs (logits[B,D], queue_anchor[K,D], queue_label[K], knn,
// n_class). Output: [B, n_class] fp32.
// ---------------------------------------------------------------------------
extern "C" void kernel_launch(void* const* d_in, const int* in_sizes, int n_in,
                              void* d_out, int out_size, void* d_ws, size_t ws_size,
                              hipStream_t stream)
{
    const float* logits = (const float*)d_in[0];
    const float* anchor = (const float*)d_in[1];
    const int*   labels = (const int*)d_in[2];
    const int*   knn_p  = (const int*)d_in[3];
    const int*   ncls_p = (const int*)d_in[4];

    const int K = in_sizes[2];                 // 4096
    const int D = in_sizes[1] / K;             // 50257
    const int B = in_sizes[0] / D;             // 128
    const int Dp = (D + 31) & ~31;             // 50272

    char* ws = (char*)d_ws;
    __bf16* xlog   = (__bf16*)ws;                          // B*Dp bf16
    size_t  xbytes = ((size_t)B * Dp * sizeof(__bf16) + 255) & ~(size_t)255;
    float*  scaled = (float*)(ws + xbytes);                // B*K fp32

    {
        long long total = (long long)B * Dp;
        int blocks = (int)((total + 255) / 256);
        log_convert_kernel<<<blocks, 256, 0, stream>>>(logits, xlog, B, D, Dp);
    }
    {
        dim3 grid((K + 15) / 16, (B + 127) / 128);
        knn_gemm_kernel<<<grid, 256, 0, stream>>>(xlog, anchor, scaled, B, D, Dp, K);
    }
    {
        topk_kernel<<<B, 256, 0, stream>>>(scaled, labels, knn_p, ncls_p,
                                           (float*)d_out, K);
    }
}
